// SimpleQueryDecoder_30219389894874
// MI455X (gfx1250) — matrix operand order repaired and checked by reading, not compiled, not run on hardware
//
#include <hip/hip_runtime.h>
#include <math.h>

// ---------------------------------------------------------------------------
// SimpleQueryDecoder, fused single-pass flash-decoding for MI455X (gfx1250).
//
//  k_prep_q : q~ = ((query_embed@Wq^T + bq) @ Wk) * inv_temp/16   -> bf16 [128,256]
//  k_stream : per (batch, N-chunk): stream memory tiles ONCE via async
//             global->LDS double buffering; online-softmax
//             scores = q~ . mem  (v_wmma_f32_16x16x32_bf16),
//             ctx += P . mem, anchor += P . norm_coords, colsum(mem), count(mask)
//  k_final  : combine chunk partials; qfeat = ctx@Wv^T+bv (WMMA); FiLM with
//             gamma/beta MLPs; cw1/sw1 hidden GEMMs (WMMA); heads; outputs.
// ---------------------------------------------------------------------------

#define D_MODEL 256
#define NQ      100
#define QP      128            // queries padded to 8 WMMA row-tiles
#define BATCH   32
#define NMEM    8192
#define NC      8              // N-chunks per batch
#define CHUNK   (NMEM / NC)    // 1024 rows
#define TN      32             // rows per streamed tile
#define NTILES  (CHUNK / TN)   // 32 tiles per chunk

typedef __bf16 bf16_t;
typedef __attribute__((ext_vector_type(16))) __bf16 v16bf;
typedef __attribute__((ext_vector_type(8)))  __bf16 v8bf;
typedef __attribute__((ext_vector_type(2)))  __bf16 v2bf;
typedef __attribute__((ext_vector_type(8)))  float  v8f;
typedef __attribute__((ext_vector_type(4)))  float  v4f;
typedef __attribute__((ext_vector_type(4)))  int    v4i;

// ---- gfx1250 async global->LDS (ASYNCcnt) --------------------------------
#if defined(__HIP_DEVICE_COMPILE__) && defined(__has_builtin)
# if __has_builtin(__builtin_amdgcn_global_load_async_to_lds_b128)
#  define HAVE_ASYNC_LDS 1
# endif
# if __has_builtin(__builtin_amdgcn_s_wait_asynccnt)
#  define HAVE_WAIT_ASYNC_BUILTIN 1
# endif
#endif

#ifdef HAVE_ASYNC_LDS
__device__ __forceinline__ void wait_async0() {
#ifdef HAVE_WAIT_ASYNC_BUILTIN
  __builtin_amdgcn_s_wait_asynccnt(0);
#else
  asm volatile("s_wait_asynccnt 0" ::: "memory");
#endif
}
// Copy one contiguous 32 KB tile (32 rows x 256 f32) global -> LDS staging.
// Per wave-instruction each lane moves 16B; 8 issues/thread covers the tile.
__device__ __forceinline__ void async_tile_copy(const float* g, float* l, int tid) {
#pragma unroll
  for (int i = 0; i < 8; ++i) {
    const float* gp = g + (i * 256 + tid) * 4;
    float* lp = l + (i * 256 + tid) * 4;
    __builtin_amdgcn_global_load_async_to_lds_b128(
        (__attribute__((address_space(1))) v4i*)(gp),
        (__attribute__((address_space(3))) v4i*)(lp), 0, 0);
  }
}
#endif

// ---- WMMA operand loaders (CDNA5 ISA 7.12.2 layouts, wave32) --------------
// A (16x32 bf16, M x K): lane m(0-15)/m+16 holds k=[8h,8h+8) and [16+8h,16+8h+8)
__device__ __forceinline__ v16bf load_a16x32(const bf16_t* base, int row0,
                                             int stride, int kbase, int lane) {
  const int m = lane & 15, h = lane >> 4;
  const bf16_t* p = base + (size_t)(row0 + m) * stride + kbase + 8 * h;
  v8bf lo = *(const v8bf*)p;
  v8bf hi = *(const v8bf*)(p + 16);
  return __builtin_shufflevector(lo, hi, 0,1,2,3,4,5,6,7,8,9,10,11,12,13,14,15);
}
// B (32x16 bf16, K x N) from an [N][K] row-major buffer:
// lane n(0-15)+16h holds k=[8h,8h+8) and [16+8h,16+8h+8) (same order as A).
__device__ __forceinline__ v16bf load_b32x16(const bf16_t* base, int col0,
                                             int stride, int kbase, int lane) {
  const bf16_t* p = base + (size_t)(col0 + (lane & 15)) * stride + kbase + 8 * (lane >> 4);
  v8bf lo = *(const v8bf*)p;
  v8bf hi = *(const v8bf*)(p + 16);
  return __builtin_shufflevector(lo, hi, 0,1,2,3,4,5,6,7,8,9,10,11,12,13,14,15);
}
__device__ __forceinline__ v8f wmma_bf16(v16bf a, v16bf b, v8f c) {
  return __builtin_amdgcn_wmma_f32_16x16x32_bf16(false, a, false, b, (short)0, c,
                                                 false, false);
}
// C tile of a 256-deep GEMM: A [QP][256] bf16, W [dout][din] bf16 (row major)
__device__ __forceinline__ v8f gemm_tile256(const bf16_t* A, const bf16_t* W,
                                            int qb, int col0, int lane) {
  v8f acc = {0.f,0.f,0.f,0.f,0.f,0.f,0.f,0.f};
#pragma unroll
  for (int kc = 0; kc < 8; ++kc)
    acc = wmma_bf16(load_a16x32(A, qb, D_MODEL, kc * 32, lane),
                    load_b32x16(W, col0, D_MODEL, kc * 32, lane), acc);
  return acc;
}

// ---------------------------------------------------------------------------
// Kernel 1: q~ = ((query_embed @ Wq^T + bq) @ Wk) * inv_temp/sqrt(256)
// (bk contributes a per-query constant to scores -> softmax invariant.)
__global__ __launch_bounds__(256) void k_prep_q(
    const float* __restrict__ qe, const float* __restrict__ Wq,
    const float* __restrict__ bq, const float* __restrict__ Wk,
    const float* __restrict__ invt, bf16_t* __restrict__ qt) {
  __shared__ float qrow[D_MODEL];
  const int qi = blockIdx.x, d = threadIdx.x;
  float acc = 0.f;
  if (qi < NQ) {
    const float* qr = qe + qi * D_MODEL;
    for (int k = 0; k < D_MODEL; ++k) acc += qr[k] * Wq[d * D_MODEL + k];
    acc += bq[d];
  }
  qrow[d] = acc;
  __syncthreads();
  const float s = invt[0] * (1.0f / 16.0f);
  float o = 0.f;
  for (int k = 0; k < D_MODEL; ++k) o += qrow[k] * Wk[k * D_MODEL + d];
  qt[qi * D_MODEL + d] = (bf16_t)(o * s);
}

// ---------------------------------------------------------------------------
// Kernel 2: streaming flash-decoding partials.  grid = BATCH*NC, block = 256.
struct Smem2 {
  float  stg[2][TN * D_MODEL];  // 64 KB  double-buffered f32 async staging
  bf16_t qt[QP * D_MODEL];      // 64 KB  q~ (A operand)
  bf16_t memB[TN * D_MODEL];    // 16 KB  tile, [n][d]  (B of scores)
  bf16_t memT[D_MODEL * TN];    // 16 KB  tile, [d][n]  (B of ctx GEMM)
  float  sc[QP * TN];           // 16 KB  raw scores
  bf16_t p[QP * TN];            //  8 KB  probabilities (A of ctx GEMM)
  float  coords[TN * 3];
  float  mask[TN];
  float  m[QP];
  float  l[QP];
  float  scl[QP];
  float  anc[QP * 3];
  float  red[TN];
};

__global__ __launch_bounds__(256) void k_stream(
    const float* __restrict__ mem, const float* __restrict__ coords,
    const float* __restrict__ mean, const float* __restrict__ scale,
    const unsigned char* __restrict__ mask, const bf16_t* __restrict__ qt,
    float* __restrict__ ctxPart, float* __restrict__ mPart,
    float* __restrict__ lPart, float* __restrict__ ancPart,
    float* __restrict__ msumPart, float* __restrict__ cntPart) {
  extern __shared__ __attribute__((aligned(128))) char smem_raw[];
  Smem2& S = *(Smem2*)smem_raw;

  const int tid = threadIdx.x, lane = tid & 31, wave = tid >> 5;
  const int bc = blockIdx.x, b = bc >> 3, chunk = bc & (NC - 1);
  const int n0 = chunk * CHUNK;
  const int qb = wave * 16, h = lane >> 4;

  { // stage q~ into LDS (16B vector copies)
    const uint4* src = (const uint4*)qt;
    uint4* dst = (uint4*)S.qt;
    for (int i = tid; i < (QP * D_MODEL * 2) / 16; i += 256) dst[i] = src[i];
  }
  if (tid < QP) {
    S.m[tid] = -1e30f; S.l[tid] = 0.f;
    S.anc[tid * 3 + 0] = S.anc[tid * 3 + 1] = S.anc[tid * 3 + 2] = 0.f;
  }
  v8f acc[16];
#pragma unroll
  for (int i = 0; i < 16; ++i)
#pragma unroll
    for (int r = 0; r < 8; ++r) acc[i][r] = 0.f;

  float msum_acc = 0.f;
  int cnt_acc = 0;
  const int row = tid >> 3, colb = (tid & 7) * 32;
  float c_mean = 0.f, c_iscale = 1.f; int cr = 0, cc = 0;
  if (tid < 96) {
    cr = tid / 3; cc = tid % 3;
    c_mean = mean[b * 3 + cc];
    c_iscale = 1.f / scale[b * 3 + cc];
  }
  __syncthreads();

#ifdef HAVE_ASYNC_LDS
  async_tile_copy(mem + ((size_t)b * NMEM + n0) * D_MODEL, S.stg[0], tid);
#endif

  for (int t = 0; t < NTILES; ++t) {
    const int nbase = n0 + t * TN;
    // small per-tile loads overlap the async wait
    const unsigned char mk = mask[(size_t)b * NMEM + nbase + row];
    float cval = 0.f;
    if (tid < 96)
      cval = (coords[((size_t)b * NMEM + nbase + cr) * 3 + cc] - c_mean) * c_iscale;

#ifdef HAVE_ASYNC_LDS
    wait_async0();           // own wave's copy of tile t complete
    __syncthreads();         // all waves' portions visible in staging
    if (t + 1 < NTILES)      // kick off tile t+1 while we compute on t
      async_tile_copy(mem + ((size_t)b * NMEM + nbase + TN) * D_MODEL,
                      S.stg[(t + 1) & 1], tid);
    const float* srcp = &S.stg[t & 1][row * D_MODEL + colb];
#else
    const float* srcp = mem + ((size_t)b * NMEM + nbase + row) * D_MODEL + colb;
    if (t + 1 < NTILES)      // global_prefetch_b8 of next tile
      __builtin_prefetch(mem + ((size_t)b * NMEM + nbase + TN + row) * D_MODEL + colb, 0, 1);
#endif

    // ---- convert tile: f32 -> bf16 LDS (row-major packed + transposed) ---
    {
      const float mkf = mk ? 1.f : 0.f;
      float vals[32];
#pragma unroll
      for (int i = 0; i < 8; ++i) {
        v4f v = *(const v4f*)(srcp + i * 4);
        vals[i*4+0] = v[0]; vals[i*4+1] = v[1]; vals[i*4+2] = v[2]; vals[i*4+3] = v[3];
      }
#pragma unroll
      for (int j = 0; j < 16; ++j) {
        bf16_t h0 = (bf16_t)(vals[2*j]     * mkf);
        bf16_t h1 = (bf16_t)(vals[2*j + 1] * mkf);
        v2bf pk; pk[0] = h0; pk[1] = h1;
        *(v2bf*)&S.memB[row * D_MODEL + colb + 2*j] = pk;   // one b32 store
        S.memT[(colb + 2*j)     * TN + row] = h0;
        S.memT[(colb + 2*j + 1) * TN + row] = h1;
      }
      if ((tid & 7) == 0) { S.mask[row] = mkf; cnt_acc += mk ? 1 : 0; }
      if (tid < 96) S.coords[tid] = cval;
    }
    __syncthreads();

    // masked column-sum partial for the global mean (thread owns col d=tid)
    {
      float s = 0.f;
#pragma unroll
      for (int i = 0; i < TN; ++i) s += (float)S.memT[tid * TN + i];
      msum_acc += s;
    }

    // ---- scores = q~ . mem_tile^T : wave qb owns 16 queries --------------
    for (int nt = 0; nt < 2; ++nt) {
      v8f c = {0.f,0.f,0.f,0.f,0.f,0.f,0.f,0.f};
#pragma unroll
      for (int kc = 0; kc < 8; ++kc)
        c = wmma_bf16(load_a16x32(S.qt, qb, D_MODEL, kc * 32, lane),
                      load_b32x16(S.memB, nt * 16, D_MODEL, kc * 32, lane), c);
      const int nn = nt * 16 + (lane & 15);
#pragma unroll
      for (int r = 0; r < 8; ++r) S.sc[(qb + r + 8 * h) * TN + nn] = c[r];
    }
    __syncthreads();

    // ---- online softmax update (2 threads per query) ---------------------
    {
      const int qi = tid >> 1, hn = tid & 1, nb2 = hn * 16;
      float sv[16];
      float mx = -1e30f;
#pragma unroll
      for (int i = 0; i < 16; ++i) {
        float s = S.sc[qi * TN + nb2 + i];
        s = (S.mask[nb2 + i] > 0.f) ? s : -1e30f;
        sv[i] = s; mx = fmaxf(mx, s);
      }
      mx = fmaxf(mx, __shfl_xor(mx, 1, 32));
      const float mold = S.m[qi];
      const float mnew = fmaxf(mold, mx);
      const float sclf = __expf(mold - mnew);
      float ssum = 0.f, a0 = 0.f, a1 = 0.f, a2 = 0.f;
#pragma unroll
      for (int i = 0; i < 16; ++i) {
        float pv = __expf(sv[i] - mnew);
        ssum += pv;
        S.p[qi * TN + nb2 + i] = (bf16_t)pv;
        a0 += pv * S.coords[(nb2 + i) * 3 + 0];
        a1 += pv * S.coords[(nb2 + i) * 3 + 1];
        a2 += pv * S.coords[(nb2 + i) * 3 + 2];
      }
      ssum += __shfl_xor(ssum, 1, 32);
      a0 += __shfl_xor(a0, 1, 32);
      a1 += __shfl_xor(a1, 1, 32);
      a2 += __shfl_xor(a2, 1, 32);
      if (hn == 0) {
        S.m[qi] = mnew;
        S.l[qi] = S.l[qi] * sclf + ssum;
        S.scl[qi] = sclf;
        S.anc[qi * 3 + 0] = S.anc[qi * 3 + 0] * sclf + a0;
        S.anc[qi * 3 + 1] = S.anc[qi * 3 + 1] * sclf + a1;
        S.anc[qi * 3 + 2] = S.anc[qi * 3 + 2] * sclf + a2;
      }
    }
    __syncthreads();

    // ---- rescale ctx accumulators + ctx += P . mem_tile ------------------
    {
      float srs[8];
#pragma unroll
      for (int r = 0; r < 8; ++r) srs[r] = S.scl[qb + r + 8 * h];
#pragma unroll
      for (int dt = 0; dt < 16; ++dt)
#pragma unroll
        for (int r = 0; r < 8; ++r) acc[dt][r] = acc[dt][r] * srs[r];
      v16bf ap = load_a16x32(S.p, qb, TN, 0, lane);  // P tile 16x32
#pragma unroll
      for (int dt = 0; dt < 16; ++dt)
        acc[dt] = wmma_bf16(ap, load_b32x16(S.memT, dt * 16, TN, 0, lane), acc[dt]);
    }
    __syncthreads();
  }

  // ---- write chunk partials ----------------------------------------------
#pragma unroll
  for (int dt = 0; dt < 16; ++dt)
#pragma unroll
    for (int r = 0; r < 8; ++r)
      ctxPart[((size_t)bc * QP + qb + r + 8 * h) * D_MODEL + dt * 16 + (lane & 15)] =
          acc[dt][r];
  if (tid < QP) {
    mPart[(size_t)bc * QP + tid] = S.m[tid];
    lPart[(size_t)bc * QP + tid] = S.l[tid];
    ancPart[((size_t)bc * QP + tid) * 3 + 0] = S.anc[tid * 3 + 0];
    ancPart[((size_t)bc * QP + tid) * 3 + 1] = S.anc[tid * 3 + 1];
    ancPart[((size_t)bc * QP + tid) * 3 + 2] = S.anc[tid * 3 + 2];
  }
  msumPart[(size_t)bc * D_MODEL + tid] = msum_acc;
  if ((tid & 7) == 0) S.red[row] = (float)cnt_acc;
  __syncthreads();
  if (tid == 0) {
    float s = 0.f;
    for (int i = 0; i < TN; ++i) s += S.red[i];
    cntPart[bc] = s;
  }
}

// ---------------------------------------------------------------------------
// Kernel 3: combine partials + epilogue GEMMs.  grid = BATCH, block = 256.
struct Smem3 {
  bf16_t x[QP * D_MODEL];          // 64 KB  ctx_n -> hidden buffers
  bf16_t dec[QP * D_MODEL];        // 64 KB  decoded
  bf16_t w[D_MODEL * D_MODEL];     // 128 KB weight staging (bf16)
  float coef[NC * QP];
  float linv[QP];
  float anc[QP * 3];
  float gf[D_MODEL];
  float h1[D_MODEL];
  float gam[D_MODEL];              // holds (1 + gamma)
  float bet[D_MODEL];
  float dc[QP * 3];
  float sz[QP * 3];
};

__global__ __launch_bounds__(256) void k_final(
    const float* __restrict__ mean, const float* __restrict__ scale,
    const float* __restrict__ Wv, const float* __restrict__ bv,
    const float* __restrict__ gw1, const float* __restrict__ gb1,
    const float* __restrict__ gw2, const float* __restrict__ gb2,
    const float* __restrict__ bw1, const float* __restrict__ bb1,
    const float* __restrict__ bw2, const float* __restrict__ bb2,
    const float* __restrict__ cw1, const float* __restrict__ cb1,
    const float* __restrict__ cw2, const float* __restrict__ cb2,
    const float* __restrict__ sw1, const float* __restrict__ sb1,
    const float* __restrict__ sw2, const float* __restrict__ sb2,
    const float* __restrict__ clw, const float* __restrict__ clb,
    const float* __restrict__ ctxPart, const float* __restrict__ mPart,
    const float* __restrict__ lPart, const float* __restrict__ ancPart,
    const float* __restrict__ msumPart, const float* __restrict__ cntPart,
    float* __restrict__ outBoxes, float* __restrict__ outClasses) {
  extern __shared__ __attribute__((aligned(128))) char smem_raw[];
  Smem3& S = *(Smem3*)smem_raw;
  const int tid = threadIdx.x, lane = tid & 31, wave = tid >> 5;
  const int b = blockIdx.x;
  const int qb = wave * 16, h = lane >> 4, nn = lane & 15;
  const size_t base = (size_t)b * NC;

  // P1: per-query softmax combination across chunks
  if (tid < QP) {
    const int q = tid;
    float M = -3.0e38f;
#pragma unroll
    for (int c = 0; c < NC; ++c) M = fmaxf(M, mPart[(base + c) * QP + q]);
    float L = 0.f;
#pragma unroll
    for (int c = 0; c < NC; ++c) {
      float w = __expf(mPart[(base + c) * QP + q] - M);
      S.coef[c * QP + q] = w;
      L += w * lPart[(base + c) * QP + q];
    }
    const float li = 1.f / fmaxf(L, 1e-30f);
    S.linv[q] = li;
    for (int k = 0; k < 3; ++k) {
      float a = 0.f;
#pragma unroll
      for (int c = 0; c < NC; ++c)
        a += S.coef[c * QP + q] * ancPart[((base + c) * QP + q) * 3 + k];
      S.anc[q * 3 + k] = a * li;
    }
  }
  __syncthreads();

  // P2: ctx_n (attn @ mem) -> bf16 x;  masked-mean global feature
  {
    float ms = 0.f, cnt = 0.f;
#pragma unroll
    for (int c = 0; c < NC; ++c) {
      ms += msumPart[(base + c) * D_MODEL + tid];
      cnt += cntPart[base + c];
    }
    S.gf[tid] = ms / fmaxf(cnt, 1.f);
  }
  for (int q = 0; q < QP; ++q) {
    float s = 0.f;
#pragma unroll
    for (int c = 0; c < NC; ++c)
      s += S.coef[c * QP + q] * ctxPart[((base + c) * QP + q) * D_MODEL + tid];
    S.x[q * D_MODEL + tid] = (bf16_t)(s * S.linv[q]);
  }
  __syncthreads();

  // P3: FiLM MLPs (thread = output channel)
  {
    float s = gb1[tid];
    for (int k = 0; k < D_MODEL; ++k) s += gw1[tid * D_MODEL + k] * S.gf[k];
    S.h1[tid] = fmaxf(s, 0.f);
  }
  __syncthreads();
  {
    float s = gb2[tid];
    for (int k = 0; k < D_MODEL; ++k) s += gw2[tid * D_MODEL + k] * S.h1[k];
    S.gam[tid] = 1.f + s;
  }
  __syncthreads();
  {
    float s = bb1[tid];
    for (int k = 0; k < D_MODEL; ++k) s += bw1[tid * D_MODEL + k] * S.gf[k];
    S.h1[tid] = fmaxf(s, 0.f);
  }
  __syncthreads();
  {
    float s = bb2[tid];
    for (int k = 0; k < D_MODEL; ++k) s += bw2[tid * D_MODEL + k] * S.h1[k];
    S.bet[tid] = s;
  }
  __syncthreads();

  // P4: decoded = (ctx_n @ Wv^T + bv) * (1+gamma) + beta        [WMMA]
  for (int i = tid; i < D_MODEL * D_MODEL; i += 256) S.w[i] = (bf16_t)Wv[i];
  __syncthreads();
  for (int dt = 0; dt < 16; ++dt) {
    v8f c = gemm_tile256(S.x, S.w, qb, dt * 16, lane);
    const int d = dt * 16 + nn;
    const float bvv = bv[d], ga = S.gam[d], be = S.bet[d];
#pragma unroll
    for (int r = 0; r < 8; ++r)
      S.dec[(qb + r + 8 * h) * D_MODEL + d] = (bf16_t)((c[r] + bvv) * ga + be);
  }
  __syncthreads();

  // P5: center-head hidden = relu(decoded @ cw1^T + cb1)        [WMMA]
  for (int i = tid; i < D_MODEL * D_MODEL; i += 256) S.w[i] = (bf16_t)cw1[i];
  __syncthreads();
  for (int dt = 0; dt < 16; ++dt) {
    v8f c = gemm_tile256(S.dec, S.w, qb, dt * 16, lane);
    const int d = dt * 16 + nn;
    const float cb = cb1[d];
#pragma unroll
    for (int r = 0; r < 8; ++r)
      S.x[(qb + r + 8 * h) * D_MODEL + d] = (bf16_t)fmaxf(c[r] + cb, 0.f);
  }
  __syncthreads();
  if (tid < QP) {
    const int q = tid;
    for (int k = 0; k < 3; ++k) {
      float s = cb2[k];
      for (int d = 0; d < D_MODEL; ++d) s += cw2[k * D_MODEL + d] * (float)S.x[q * D_MODEL + d];
      S.dc[q * 3 + k] = s;
    }
  }
  __syncthreads();

  // P6: size-head hidden = relu(decoded @ sw1^T + sb1)          [WMMA]
  for (int i = tid; i < D_MODEL * D_MODEL; i += 256) S.w[i] = (bf16_t)sw1[i];
  __syncthreads();
  for (int dt = 0; dt < 16; ++dt) {
    v8f c = gemm_tile256(S.dec, S.w, qb, dt * 16, lane);
    const int d = dt * 16 + nn;
    const float sb = sb1[d];
#pragma unroll
    for (int r = 0; r < 8; ++r)
      S.x[(qb + r + 8 * h) * D_MODEL + d] = (bf16_t)fmaxf(c[r] + sb, 0.f);
  }
  __syncthreads();

  // P7: heads + outputs (only real queries q < 100)
  if (tid < NQ) {
    const int q = tid;
    for (int k = 0; k < 3; ++k) {
      float s = sb2[k];
      for (int d = 0; d < D_MODEL; ++d) s += sw2[k * D_MODEL + d] * (float)S.x[q * D_MODEL + d];
      const float sp = (s > 20.f) ? s : log1pf(__expf(s));
      S.sz[q * 3 + k] = sp + 1e-4f;
    }
    for (int k = 0; k < 4; ++k) {
      float s = clb[k];
      for (int d = 0; d < D_MODEL; ++d) s += clw[k * D_MODEL + d] * (float)S.dec[q * D_MODEL + d];
      outClasses[((size_t)b * NQ + q) * 4 + k] = s;
    }
    for (int k = 0; k < 3; ++k) {
      const float sc = scale[b * 3 + k], mn = mean[b * 3 + k];
      outBoxes[((size_t)b * NQ + q) * 6 + k] =
          (S.anc[q * 3 + k] + S.dc[q * 3 + k]) * sc + mn;
      outBoxes[((size_t)b * NQ + q) * 6 + 3 + k] = S.sz[q * 3 + k] * sc;
    }
  }
}

// ---------------------------------------------------------------------------
extern "C" void kernel_launch(void* const* d_in, const int* in_sizes, int n_in,
                              void* d_out, int out_size, void* d_ws, size_t ws_size,
                              hipStream_t stream) {
  (void)in_sizes; (void)n_in; (void)out_size; (void)ws_size;
  const float* memory = (const float*)d_in[0];
  const float* coords = (const float*)d_in[1];
  const float* meanp  = (const float*)d_in[2];
  const float* scalep = (const float*)d_in[3];
  const unsigned char* maskp = (const unsigned char*)d_in[4];
  const float* qe  = (const float*)d_in[5];
  const float* Wq  = (const float*)d_in[6];
  const float* bq  = (const float*)d_in[7];
  const float* Wk  = (const float*)d_in[8];
  /* bk = d_in[9]: per-query constant in scores -> softmax invariant, unused */
  const float* Wv  = (const float*)d_in[10];
  const float* bv  = (const float*)d_in[11];
  const float* gw1 = (const float*)d_in[12]; const float* gb1 = (const float*)d_in[13];
  const float* gw2 = (const float*)d_in[14]; const float* gb2 = (const float*)d_in[15];
  const float* bw1 = (const float*)d_in[16]; const float* bb1 = (const float*)d_in[17];
  const float* bw2 = (const float*)d_in[18]; const float* bb2 = (const float*)d_in[19];
  const float* cw1 = (const float*)d_in[20]; const float* cb1 = (const float*)d_in[21];
  const float* cw2 = (const float*)d_in[22]; const float* cb2 = (const float*)d_in[23];
  const float* sw1 = (const float*)d_in[24]; const float* sb1 = (const float*)d_in[25];
  const float* sw2 = (const float*)d_in[26]; const float* sb2 = (const float*)d_in[27];
  const float* clw = (const float*)d_in[28]; const float* clb = (const float*)d_in[29];
  const float* invt = (const float*)d_in[30];

  // Workspace layout (~34.6 MB)
  char* ws = (char*)d_ws;
  const size_t NCHT = (size_t)BATCH * NC;
  size_t off = 0;
  bf16_t* qt = (bf16_t*)(ws + off);      off += (size_t)QP * D_MODEL * 2;
  off = (off + 255) & ~(size_t)255;
  float* ctxPart = (float*)(ws + off);   off += NCHT * QP * D_MODEL * 4;
  float* mPart   = (float*)(ws + off);   off += NCHT * QP * 4;
  float* lPart   = (float*)(ws + off);   off += NCHT * QP * 4;
  float* ancPart = (float*)(ws + off);   off += NCHT * QP * 3 * 4;
  float* msumPart= (float*)(ws + off);   off += NCHT * D_MODEL * 4;
  float* cntPart = (float*)(ws + off);   off += NCHT * 4;

  float* outBoxes = (float*)d_out;                        // [B,100,6]
  float* outClasses = outBoxes + (size_t)BATCH * NQ * 6;  // [B,100,4]

  k_prep_q<<<QP, 256, 0, stream>>>(qe, Wq, bq, Wk, invt, qt);
  k_stream<<<BATCH * NC, 256, sizeof(Smem2), stream>>>(
      memory, coords, meanp, scalep, maskp, qt,
      ctxPart, mPart, lPart, ancPart, msumPart, cntPart);
  k_final<<<BATCH, 256, sizeof(Smem3), stream>>>(
      meanp, scalep, Wv, bv, gw1, gb1, gw2, gb2, bw1, bb1, bw2, bb2,
      cw1, cb1, cw2, cb2, sw1, sb1, sw2, sb2, clw, clb,
      ctxPart, mPart, lPart, ancPart, msumPart, cntPart,
      outBoxes, outClasses);
}
